// PRNNCell_70428873720509
// MI455X (gfx1250) — matrix operand order repaired and checked by reading, not compiled, not run on hardware
//
#include <hip/hip_runtime.h>
#include <hip/hip_bf16.h>
#include <math.h>

#define ALPHA 0.001f
constexpr int NB = 64;            // batch
constexpr int NI = 512;           // input dim
constexpr int NH = 1024;          // hidden dim
constexpr int ND = NI + NH;       // 1536 concat dim

typedef __attribute__((ext_vector_type(16))) _Float16 v16h;
typedef __attribute__((ext_vector_type(8)))  float    v8f;

__device__ __forceinline__ float4 ldg4(const float* p) {
    return *reinterpret_cast<const float4*>(p);
}

// ---------------------------------------------------------------------------
// Fused kernel: one wave32 per (b,k) row of wih.
//   phase 1: y = tanh( dot(wih[b,k,:], x[b,:]) + bih[k] )   (cross-lane reduce)
//   phase 2: wih_new[b,k,j] = wih + a*( x*(y*A + B) + (y*C + D) )
// wih row is 6KB: phase-2 re-read hits WGP$/L2, so HBM sees wih exactly once.
// ---------------------------------------------------------------------------
__global__ __launch_bounds__(256)
void prnn_fused_kernel(const float* __restrict__ inp, const float* __restrict__ hid,
                       const float* __restrict__ wih,
                       const float* __restrict__ Wa, const float* __restrict__ Wb,
                       const float* __restrict__ Wc, const float* __restrict__ Wd,
                       const float* __restrict__ bih,
                       float* __restrict__ out_h, float* __restrict__ out_w)
{
    const int lane = threadIdx.x & 31;
    const int wave = threadIdx.x >> 5;
    const int row  = blockIdx.x * (blockDim.x >> 5) + wave;   // row = b*NH + k
    const int b    = row >> 10;                               // NH == 1024
    const int k    = row & (NH - 1);

    const size_t roff = (size_t)row * ND;
    const float* wrow = wih + roff;
    const float* xin  = inp + (size_t)b * NI;
    const float* xhid = hid + (size_t)b * NH;

    // ---- phase 1: dot(wih_row, x) ----
    float acc = 0.f;
    #pragma unroll 4
    for (int c = lane; c < ND / 4; c += 32) {
        const int j0 = c * 4;                                 // 16B aligned; NI%4==0
        float4 w4 = ldg4(wrow + j0);
        float4 x4 = (j0 < NI) ? ldg4(xin + j0) : ldg4(xhid + (j0 - NI));
        acc += w4.x * x4.x + w4.y * x4.y + w4.z * x4.z + w4.w * x4.w;
    }
    #pragma unroll
    for (int off = 16; off > 0; off >>= 1)
        acc += __shfl_xor(acc, off, 32);                      // wave32 butterfly reduce
    const float y = tanhf(acc + bih[k]);
    if (lane == 0) out_h[row] = y;

    // ---- phase 2: plastic update, streamed ----
    const float* arow = Wa + (size_t)k * ND;
    const float* brow = Wb + (size_t)k * ND;
    const float* crow = Wc + (size_t)k * ND;
    const float* drow = Wd + (size_t)k * ND;
    float*       orow = out_w + roff;

    #pragma unroll 2
    for (int c = lane; c < ND / 4; c += 32) {
        const int j0 = c * 4;
        __builtin_prefetch(arow + j0 + 512, 0, 0);            // global_prefetch_b8
        float4 w4 = ldg4(wrow + j0);
        float4 x4 = (j0 < NI) ? ldg4(xin + j0) : ldg4(xhid + (j0 - NI));
        float4 a4 = ldg4(arow + j0);
        float4 b4 = ldg4(brow + j0);
        float4 c4 = ldg4(crow + j0);
        float4 d4 = ldg4(drow + j0);
        float4 r;
        r.x = fmaf(ALPHA, fmaf(x4.x, fmaf(y, a4.x, b4.x), fmaf(y, c4.x, d4.x)), w4.x);
        r.y = fmaf(ALPHA, fmaf(x4.y, fmaf(y, a4.y, b4.y), fmaf(y, c4.y, d4.y)), w4.y);
        r.z = fmaf(ALPHA, fmaf(x4.z, fmaf(y, a4.z, b4.z), fmaf(y, c4.z, d4.z)), w4.z);
        r.w = fmaf(ALPHA, fmaf(x4.w, fmaf(y, a4.w, b4.w), fmaf(y, c4.w, d4.w)), w4.w);
        *reinterpret_cast<float4*>(orow + j0) = r;
    }
}

// ---------------------------------------------------------------------------
// Dopamine GEMM (reference computes sigmoid(x @ W_dop.T + b_dop)):
// v_wmma_f32_16x16x32_f16, one wave per 16x16 output tile.
//   A (16x32, f16): lane m=l&15, h=l>>4 holds K in [k0+8h, k0+8h+8) and
//                   [k0+16+8h, k0+16+8h+8)        (ISA 16-bit A-matrix layout)
//   B (32x16, f16): lane col n=l&15, h=l>>4 holds K in [k0+8h, k0+8h+8), [k0+16+8h, k0+16+8h+8)
//   D (16x16, f32): D[v][lane] = elem(M = v + 8h, N = l&15)
// Output -> scratch (reference's _dopamine is not a returned output).
// ---------------------------------------------------------------------------
__global__ __launch_bounds__(32)
void dopamine_wmma_kernel(const float* __restrict__ inp, const float* __restrict__ hid,
                          const float* __restrict__ Wdop, const float* __restrict__ bdop,
                          float* __restrict__ dop)
{
    const int lane = threadIdx.x & 31;
    const int m    = lane & 15;
    const int h    = lane >> 4;
    const int btile = blockIdx.x;                 // 0..NB/16-1
    const int ntile = blockIdx.y;                 // 0..NH/16-1
    const int brow  = btile * 16 + m;             // batch row for A
    const int ncol  = ntile * 16 + m;             // neuron col for B

    const float* xin  = inp  + (size_t)brow * NI;
    const float* xhid = hid  + (size_t)brow * NH;
    const float* wrow = Wdop + (size_t)ncol * ND; // W_dop.T column = W_dop row

    v8f c = {};
    for (int k0 = 0; k0 < ND; k0 += 32) {
        v16h a, bm;
        // A operand: two 8-wide segments (8-aligned, never straddle NI=512)
        {
            const int s0 = k0 + 8 * h;
            const int s1 = k0 + 16 + 8 * h;
            const float* p0 = (s0 < NI) ? (xin + s0) : (xhid + (s0 - NI));
            const float* p1 = (s1 < NI) ? (xin + s1) : (xhid + (s1 - NI));
            float4 f0 = ldg4(p0), f1 = ldg4(p0 + 4);
            float4 f2 = ldg4(p1), f3 = ldg4(p1 + 4);
            a[0]  = (_Float16)f0.x; a[1]  = (_Float16)f0.y; a[2]  = (_Float16)f0.z; a[3]  = (_Float16)f0.w;
            a[4]  = (_Float16)f1.x; a[5]  = (_Float16)f1.y; a[6]  = (_Float16)f1.z; a[7]  = (_Float16)f1.w;
            a[8]  = (_Float16)f2.x; a[9]  = (_Float16)f2.y; a[10] = (_Float16)f2.z; a[11] = (_Float16)f2.w;
            a[12] = (_Float16)f3.x; a[13] = (_Float16)f3.y; a[14] = (_Float16)f3.z; a[15] = (_Float16)f3.w;
        }
        // B operand: two 8-wide K segments from W_dop row (same order as A)
        {
            const float* p = wrow + k0 + 8 * h;
            float4 f0 = ldg4(p), f1 = ldg4(p + 4), f2 = ldg4(p + 16), f3 = ldg4(p + 20);
            bm[0]  = (_Float16)f0.x; bm[1]  = (_Float16)f0.y; bm[2]  = (_Float16)f0.z; bm[3]  = (_Float16)f0.w;
            bm[4]  = (_Float16)f1.x; bm[5]  = (_Float16)f1.y; bm[6]  = (_Float16)f1.z; bm[7]  = (_Float16)f1.w;
            bm[8]  = (_Float16)f2.x; bm[9]  = (_Float16)f2.y; bm[10] = (_Float16)f2.z; bm[11] = (_Float16)f2.w;
            bm[12] = (_Float16)f3.x; bm[13] = (_Float16)f3.y; bm[14] = (_Float16)f3.z; bm[15] = (_Float16)f3.w;
        }
        // (neg_a, A, neg_b, B, c_mod, C, reuse_a, reuse_b)
        c = __builtin_amdgcn_wmma_f32_16x16x32_f16(false, a, false, bm, (short)0, c,
                                                   false, false);
    }

    const float bias = bdop[ncol];
    #pragma unroll
    for (int v = 0; v < 8; ++v) {
        const int M = v + 8 * h;                  // batch within tile
        const int N = m;                          // neuron within tile
        float s = c[v] + bias;
        s = 1.f / (1.f + expf(-s));
        dop[(size_t)(btile * 16 + M) * NH + (ntile * 16 + N)] = s;
    }
}

extern "C" void kernel_launch(void* const* d_in, const int* in_sizes, int n_in,
                              void* d_out, int out_size, void* d_ws, size_t ws_size,
                              hipStream_t stream) {
    const float* inp  = (const float*)d_in[0];
    const float* hid  = (const float*)d_in[1];
    const float* wih  = (const float*)d_in[2];
    const float* Wa   = (const float*)d_in[3];
    const float* Wb   = (const float*)d_in[4];
    const float* Wc   = (const float*)d_in[5];
    const float* Wd   = (const float*)d_in[6];
    const float* bih  = (const float*)d_in[7];
    const float* Wdop = (const float*)d_in[8];
    const float* bdop = (const float*)d_in[9];

    float* out_h = (float*)d_out;                       // (B, H) new_hidden
    float* out_w = out_h + (size_t)NB * NH;             // (B, H, D) wih_new

    // Dopamine GEMM (WMMA) -> scratch; computed by the reference, not returned.
    if (ws_size >= (size_t)NB * NH * sizeof(float)) {
        dopamine_wmma_kernel<<<dim3(NB / 16, NH / 16), 32, 0, stream>>>(
            inp, hid, Wdop, bdop, (float*)d_ws);
    }

    // Fused hidden-update + plastic weight update: 1 wave per wih row.
    const int waves_per_block = 8;                      // 256 threads
    const int rows = NB * NH;                           // 65536 rows
    prnn_fused_kernel<<<rows / waves_per_block, waves_per_block * 32, 0, stream>>>(
        inp, hid, wih, Wa, Wb, Wc, Wd, bih, out_h, out_w);
}
